// SimilarGCN_87917980549713
// MI455X (gfx1250) — matrix operand order repaired and checked by reading, not compiled, not run on hardware
//
#include <hip/hip_runtime.h>
#include <hip/hip_bf16.h>
#include <math.h>

#define N_   8
#define C_   256
#define HW_  961
#define M_   (N_*HW_)      // 7688
#define MP_  7696          // padded to multiple of 16
#define HID_ 512
#define OUT_ 256
#define MTILES_ ((M_ + 15) / 16)   // 481

typedef __bf16 bf16x8  __attribute__((ext_vector_type(8)));
typedef __bf16 bf16x16 __attribute__((ext_vector_type(16)));
typedef float  f32x8   __attribute__((ext_vector_type(8)));

__device__ __forceinline__ float dvn(int n) {
    // D^-1/2 entries: deg = 963 on the edge samples, 964 in the middle
    return (n == 0 || n == N_ - 1) ? (1.0f / sqrtf(963.0f)) : (1.0f / sqrtf(964.0f));
}
__device__ __forceinline__ float lrelu(float x) { return x > 0.0f ? x : 0.01f * x; }

// ---- fp32 -> bf16 hi/lo split (bf16x3 GEMM emulation) ----
__global__ void k_split(const float* __restrict__ src, __bf16* __restrict__ h,
                        __bf16* __restrict__ l, int n) {
    int i = blockIdx.x * blockDim.x + threadIdx.x;
    if (i >= n) return;
    float x = src[i];
    __bf16 hh = (__bf16)x;
    h[i] = hh;
    l[i] = (__bf16)(x - (float)hh);
}

// ---- S1[n][c] = dv[n] * sum_p x[n][c][p]; one wave per (n,c), contiguous 961 floats ----
__global__ void k_blocksum1(const float* __restrict__ x, float* __restrict__ S1) {
    int wave = (blockIdx.x * blockDim.x + threadIdx.x) >> 5;
    int lane = threadIdx.x & 31;
    if (wave >= N_ * C_) return;
    const float* src = x + (size_t)wave * HW_;
    float s = 0.f;
    for (int i = lane; i < HW_; i += 32) s += src[i];
    for (int off = 16; off > 0; off >>= 1) s += __shfl_xor(s, off, 32);
    if (lane == 0) S1[wave] = dvn(wave / C_) * s;
}

// ---- G1[m][c] = dv[n]*(S1 + dv[n]*x[n,c,p] + dv[n-1]*x[n-1,c,p] + dv[n+1]*x[n+1,c,p]) ----
__global__ void k_combine1(const float* __restrict__ x, const float* __restrict__ S1,
                           __bf16* __restrict__ Gh, __bf16* __restrict__ Gl) {
    int tid = blockIdx.x * blockDim.x + threadIdx.x;
    if (tid >= N_ * C_ * HW_) return;
    int p = tid % HW_;
    int t = tid / HW_;
    int c = t % C_;
    int n = t / C_;
    float dv  = dvn(n);
    float acc = S1[n * C_ + c] + dv * x[tid];
    if (n > 0)      acc += dvn(n - 1) * x[tid - C_ * HW_];
    if (n < N_ - 1) acc += dvn(n + 1) * x[tid + C_ * HW_];
    float g = dv * acc;
    size_t o = (size_t)(n * HW_ + p) * C_ + c;   // node-major [M, C]
    __bf16 hh = (__bf16)g;
    Gh[o] = hh;
    Gl[o] = (__bf16)(g - (float)hh);
}

// ---- WMMA GEMM: C[m][nf] = lrelu(sum_k A[m][k]*W[nf][k] + bias[nf]), bf16x3 ----
// A: [MP_, K] bf16 hi/lo (row-major). B: torch-layout weight [Nf, K] bf16 hi/lo.
__global__ void k_gemm_lrelu(const __bf16* __restrict__ Ah, const __bf16* __restrict__ Al,
                             const __bf16* __restrict__ Bh, const __bf16* __restrict__ Bl,
                             const float*  __restrict__ bias, float* __restrict__ Cout,
                             int Mrows, int K, int Nf) {
    int lane = threadIdx.x & 31;
    int mt   = blockIdx.y * (blockDim.x >> 5) + (threadIdx.x >> 5);
    if (mt * 16 >= Mrows) return;                // uniform per wave
    int nt  = blockIdx.x;
    int rlo = lane & 15;
    int hi  = lane >> 4;                         // lane-half selects K sub-chunk
    const __bf16* arh = Ah + (size_t)(mt * 16 + rlo) * K;   // A row (pad rows exist, harmless)
    const __bf16* arl = Al + (size_t)(mt * 16 + rlo) * K;
    int bcol = nt * 16 + rlo;
    const __bf16* brh = Bh + (size_t)bcol * K;   // B column == contiguous W row
    const __bf16* brl = Bl + (size_t)bcol * K;

    f32x8 acc = {0.f, 0.f, 0.f, 0.f, 0.f, 0.f, 0.f, 0.f};
    for (int kt = 0; kt < K; kt += 32) {
        // A 16x32 layout: lane-half hi -> K chunks [hi*8, +8) and [16+hi*8, +8)
        bf16x8 ah0 = *(const bf16x8*)(arh + kt + hi * 8);
        bf16x8 ah1 = *(const bf16x8*)(arh + kt + 16 + hi * 8);
        bf16x8 al0 = *(const bf16x8*)(arl + kt + hi * 8);
        bf16x8 al1 = *(const bf16x8*)(arl + kt + 16 + hi * 8);
        bf16x16 a_h = __builtin_shufflevector(ah0, ah1, 0,1,2,3,4,5,6,7,8,9,10,11,12,13,14,15);
        bf16x16 a_l = __builtin_shufflevector(al0, al1, 0,1,2,3,4,5,6,7,8,9,10,11,12,13,14,15);
        // B 32x16 layout: lane-half hi -> K chunks [hi*8, +8) and [16+hi*8, +8), matching A
        bf16x8 bh0 = *(const bf16x8*)(brh + kt + hi * 8);
        bf16x8 bh1 = *(const bf16x8*)(brh + kt + 16 + hi * 8);
        bf16x8 bl0 = *(const bf16x8*)(brl + kt + hi * 8);
        bf16x8 bl1 = *(const bf16x8*)(brl + kt + 16 + hi * 8);
        bf16x16 b_h = __builtin_shufflevector(bh0, bh1, 0,1,2,3,4,5,6,7,8,9,10,11,12,13,14,15);
        bf16x16 b_l = __builtin_shufflevector(bl0, bl1, 0,1,2,3,4,5,6,7,8,9,10,11,12,13,14,15);
        acc = __builtin_amdgcn_wmma_f32_16x16x32_bf16(false, a_h, false, b_h, (short)0, acc, false, false);
        acc = __builtin_amdgcn_wmma_f32_16x16x32_bf16(false, a_h, false, b_l, (short)0, acc, false, false);
        acc = __builtin_amdgcn_wmma_f32_16x16x32_bf16(false, a_l, false, b_h, (short)0, acc, false, false);
    }
    float bv = bias[bcol];
    #pragma unroll
    for (int r = 0; r < 8; ++r) {
        int m = mt * 16 + hi * 8 + r;            // C/D layout: reg r -> row r (+8 for hi half)
        if (m < Mrows) Cout[(size_t)m * Nf + bcol] = lrelu(acc[r] + bv);
    }
}

// ---- S2[n][f] = dv[n] * sum_p H1[n*961+p][f]; fully coalesced column sums ----
__global__ void k_blocksum2(const float* __restrict__ H1, float* __restrict__ S2) {
    int n = blockIdx.x;
    int f = threadIdx.x;                         // blockDim == HID_
    const float* base = H1 + (size_t)(n * HW_) * HID_ + f;
    float s = 0.f;
    for (int p = 0; p < HW_; ++p) s += base[(size_t)p * HID_];
    S2[n * HID_ + f] = dvn(n) * s;
}

// ---- G2 = P(H1), emitted as bf16 hi/lo ----
__global__ void k_combine2(const float* __restrict__ H1, const float* __restrict__ S2,
                           __bf16* __restrict__ Gh, __bf16* __restrict__ Gl) {
    int tid = blockIdx.x * blockDim.x + threadIdx.x;
    if (tid >= M_ * HID_) return;
    int f = tid % HID_;
    int m = tid / HID_;
    int n = m / HW_;
    float dv  = dvn(n);
    float acc = S2[n * HID_ + f] + dv * H1[tid];
    if (n > 0)      acc += dvn(n - 1) * H1[tid - HW_ * HID_];
    if (n < N_ - 1) acc += dvn(n + 1) * H1[tid + HW_ * HID_];
    float g = dv * acc;
    __bf16 hh = (__bf16)g;
    Gh[tid] = hh;
    Gl[tid] = (__bf16)(g - (float)hh);
}

// ---- out[o][p] = max_n O2[n*961+p][o] ----
__global__ void k_maxred(const float* __restrict__ O2, float* __restrict__ out) {
    int tid = blockIdx.x * blockDim.x + threadIdx.x;
    if (tid >= OUT_ * HW_) return;
    int p = tid % HW_;
    int o = tid / HW_;
    float v = -INFINITY;
    #pragma unroll
    for (int n = 0; n < N_; ++n)
        v = fmaxf(v, O2[(size_t)(n * HW_ + p) * OUT_ + o]);
    out[tid] = v;
}

extern "C" void kernel_launch(void* const* d_in, const int* in_sizes, int n_in,
                              void* d_out, int out_size, void* d_ws, size_t ws_size,
                              hipStream_t stream) {
    const float* x  = (const float*)d_in[0];
    const float* W1 = (const float*)d_in[1];
    const float* b1 = (const float*)d_in[2];
    const float* W2 = (const float*)d_in[3];
    const float* b2 = (const float*)d_in[4];
    float* out = (float*)d_out;

    char* p = (char*)d_ws;
    auto carve = [&](size_t bytes) -> void* {
        void* r = (void*)p;
        p += (bytes + 255) & ~(size_t)255;
        return r;
    };
    __bf16* W1h = (__bf16*)carve((size_t)HID_ * C_ * 2);
    __bf16* W1l = (__bf16*)carve((size_t)HID_ * C_ * 2);
    __bf16* W2h = (__bf16*)carve((size_t)OUT_ * HID_ * 2);
    __bf16* W2l = (__bf16*)carve((size_t)OUT_ * HID_ * 2);
    float*  S1  = (float*) carve((size_t)N_ * C_ * 4);
    __bf16* G1h = (__bf16*)carve((size_t)MP_ * C_ * 2);
    __bf16* G1l = (__bf16*)carve((size_t)MP_ * C_ * 2);
    float*  H1  = (float*) carve((size_t)MP_ * HID_ * 4);
    float*  S2  = (float*) carve((size_t)N_ * HID_ * 4);
    __bf16* G2h = (__bf16*)carve((size_t)MP_ * HID_ * 2);
    __bf16* G2l = (__bf16*)carve((size_t)MP_ * HID_ * 2);
    float*  O2  = (float*) carve((size_t)MP_ * OUT_ * 4);

    // weight precision split
    k_split<<<(HID_ * C_ + 255) / 256, 256, 0, stream>>>(W1, W1h, W1l, HID_ * C_);
    k_split<<<(OUT_ * HID_ + 255) / 256, 256, 0, stream>>>(W2, W2h, W2l, OUT_ * HID_);

    // G1 = normalized-adjacency propagate of the NCHW input (structure-exploited)
    k_blocksum1<<<(N_ * C_ * 32 + 255) / 256, 256, 0, stream>>>(x, S1);
    k_combine1<<<(N_ * C_ * HW_ + 255) / 256, 256, 0, stream>>>(x, S1, G1h, G1l);

    // H1 = lrelu(G1 @ W1^T + b1)  [7688 x 512] via v_wmma_f32_16x16x32_bf16 (bf16x3)
    {
        dim3 grid(HID_ / 16, (MTILES_ + 7) / 8);
        k_gemm_lrelu<<<grid, 256, 0, stream>>>(G1h, G1l, W1h, W1l, b1, H1, M_, C_, HID_);
    }

    // G2 = propagate(H1)
    k_blocksum2<<<N_, HID_, 0, stream>>>(H1, S2);
    k_combine2<<<(M_ * HID_ + 255) / 256, 256, 0, stream>>>(H1, S2, G2h, G2l);

    // O2 = lrelu(G2 @ W2^T + b2)  [7688 x 256]
    {
        dim3 grid(OUT_ / 16, (MTILES_ + 7) / 8);
        k_gemm_lrelu<<<grid, 256, 0, stream>>>(G2h, G2l, W2h, W2l, b2, O2, M_, HID_, OUT_);
    }

    // max over the N=8 samples -> [1, 256, 31, 31]
    k_maxred<<<(OUT_ * HW_ + 255) / 256, 256, 0, stream>>>(O2, out);
}
